// CPAMDec_Mix_67061619360324
// MI455X (gfx1250) — matrix operand order repaired and checked by reading, not compiled, not run on hardware
//
#include <hip/hip_runtime.h>

// ---------------- CDNA5 WMMA types ----------------
typedef __attribute__((ext_vector_type(16))) __bf16 v16bf;
typedef __attribute__((ext_vector_type(8)))  float  v8f;

#define B_  8
#define C_  512
#define C4_ 128
#define K_  64
#define N_  4096
#define NT  64          // n-tile per workgroup

__device__ __forceinline__ v8f wmma_bf16(v16bf a, v16bf b, v8f c) {
  // v_wmma_f32_16x16x32_bf16  (neg_a, A, neg_b, B, c_mod, C, reuse_a, reuse_b)
  return __builtin_amdgcn_wmma_f32_16x16x32_bf16(false, a, false, b, (short)0, c, false, false);
}

__device__ __forceinline__ uint32_t lds_off(const void* p) {
  // LDS flat aperture: low 32 bits of the generic pointer are the LDS byte offset
  return (uint32_t)(uintptr_t)p;
}

// A-fragment (16x32, 16-bit) from row-major [row*ld + col].
// Lanes 0-15 row=lane, K 0..15; lanes 16-31 K 16..31 (contiguous, matches B loaders).
__device__ __forceinline__ v16bf load_a_bf16(const __bf16* base, int ld) {
  const int lane = threadIdx.x & 31;
  const int row = lane & 15, hi = lane >> 4;
  const __bf16* p = base + row * ld + hi * 16;
  v16bf a;
#pragma unroll
  for (int i = 0; i < 8; ++i) a[i] = p[i];
#pragma unroll
  for (int i = 0; i < 8; ++i) a[8 + i] = p[8 + i];
  return a;
}

__device__ __forceinline__ v16bf load_a_f32(const float* base, int ld) {
  const int lane = threadIdx.x & 31;
  const int row = lane & 15, hi = lane >> 4;
  const float* p = base + row * ld + hi * 16;
  v16bf a;
#pragma unroll
  for (int i = 0; i < 8; ++i) a[i] = (__bf16)p[i];
#pragma unroll
  for (int i = 0; i < 8; ++i) a[8 + i] = (__bf16)p[8 + i];
  return a;
}

// B-fragment (32x16) from TRANSPOSED storage bt[n][k] (k contiguous per lane).
// Lane holds column n=lane&15; lanes 0-15 K=0..15, lanes 16-31 K=16..31 (contiguous, matches A).
template <typename T>
__device__ __forceinline__ v16bf load_bt(const T* bt, int ld) {
  const int lane = threadIdx.x & 31;
  const int n = lane & 15, hi = lane >> 4;
  const T* p = bt + n * ld + hi * 16;
  v16bf b;
#pragma unroll
  for (int i = 0; i < 16; ++i) b[i] = (__bf16)p[i];
  return b;
}

// B-fragment from row-major [k][n] f32 storage (strided along k).
__device__ __forceinline__ v16bf load_b_rm_f32(const float* base, int ld) {
  const int lane = threadIdx.x & 31;
  const int n = lane & 15, hi = lane >> 4;
  const float* p = base + n + hi * 16 * ld;
  v16bf b;
#pragma unroll
  for (int i = 0; i < 16; ++i) b[i] = (__bf16)p[i * ld];
  return b;
}

// B-fragment from an LDS-staged f32 chunk [32 rows][64 cols] (rows = contraction).
__device__ __forceinline__ v16bf load_b_stage(const float* buf, int ncol0) {
  const int lane = threadIdx.x & 31;
  const int n = lane & 15, hi = lane >> 4;
  const float* p = buf + (hi * 16) * 64 + ncol0 + n;
  v16bf b;
#pragma unroll
  for (int i = 0; i < 16; ++i) b[i] = (__bf16)p[i * 64];
  return b;
}

// ---------------- stage 1: kk = y2 @ wk^T + bk  (f32, tiny) ----------------
__global__ void kk_kernel(const float* __restrict__ y2, const float* __restrict__ wk,
                          const float* __restrict__ bk, float* __restrict__ kkf) {
  const int idx = blockIdx.x * 256 + threadIdx.x;        // flat over [B][K][C4]
  const int d = idx & (C4_ - 1);
  const int k = (idx >> 7) & (K_ - 1);
  const int b = idx >> 13;
  const float4* y = (const float4*)(y2 + (size_t)(b * K_ + k) * C_);
  const float4* w = (const float4*)(wk + (size_t)d * C_);
  float acc = 0.f;
#pragma unroll 4
  for (int c = 0; c < C_ / 4; ++c) {
    float4 a = y[c], v = w[c];
    acc = fmaf(a.x, v.x, acc); acc = fmaf(a.y, v.y, acc);
    acc = fmaf(a.z, v.z, acc); acc = fmaf(a.w, v.w, acc);
  }
  kkf[idx] = acc + bk[d];
}

// ---------------- stage 2: pv{1,2} = wv @ y^T + bv  -> bf16 [b][c][k] ----------------
__global__ __launch_bounds__(256) void pv_kernel(
    const float* __restrict__ y1, const float* __restrict__ y2,
    const float* __restrict__ wv, const float* __restrict__ bv,
    __bf16* __restrict__ pv1b, __bf16* __restrict__ pv2b) {
  const int b = blockIdx.x;
  const int sel = blockIdx.y;
  const float* y = (sel ? y2 : y1) + (size_t)b * K_ * C_;
  __bf16* out = (sel ? pv2b : pv1b) + (size_t)b * C_ * K_;
  const int w = threadIdx.x >> 5;
  const int lane = threadIdx.x & 31, hi = lane >> 4, nn = lane & 15;

  for (int dt = 0; dt < 4; ++dt) {
    const int d0 = w * 64 + dt * 16;
    for (int kt = 0; kt < 4; ++kt) {
      const int k0 = kt * 16;
      v8f acc = {};
      for (int s = 0; s < C_ / 32; ++s) {
        v16bf A = load_a_f32(wv + (size_t)d0 * C_ + s * 32, C_);     // wv[d][c] row-major
        v16bf Bf = load_bt<float>(y + (size_t)k0 * C_ + s * 32, C_); // y[k][c] == (y^T) transposed
        acc = wmma_bf16(A, Bf, acc);
      }
#pragma unroll
      for (int r = 0; r < 8; ++r) {
        const int m = r + hi * 8;
        out[(size_t)(d0 + m) * K_ + k0 + nn] = (__bf16)(acc[r] + bv[d0 + m]);
      }
    }
  }
}

// ---------------- stage 3: mt[b][k][c] = kk @ wq,  ebias[b][k] = bq . kk[b][k] ----------------
__global__ __launch_bounds__(256) void m_kernel(
    const float* __restrict__ kkf, const float* __restrict__ wq,
    const float* __restrict__ bq, __bf16* __restrict__ mt,
    float* __restrict__ ebias) {
  const int b = blockIdx.x;
  const int t = threadIdx.x;
  if (t < K_) {
    const float* kr = kkf + (size_t)(b * K_ + t) * C4_;
    float e = 0.f;
#pragma unroll 4
    for (int d = 0; d < C4_; ++d) e = fmaf(bq[d], kr[d], e);
    ebias[b * K_ + t] = e;
  }
  const int w = t >> 5, lane = t & 31, hi = lane >> 4, nn = lane & 15;
  for (int kt = 0; kt < 4; ++kt) {
    const int k0 = kt * 16;
    for (int cn = 0; cn < 4; ++cn) {
      const int c0 = (w * 4 + cn) * 16;
      v8f acc = {};
      for (int s = 0; s < C4_ / 32; ++s) {
        v16bf A = load_a_f32(kkf + (size_t)(b * K_ + k0) * C4_ + s * 32, C4_); // kk[k][d]
        v16bf Bf = load_b_rm_f32(wq + (size_t)(s * 32) * C_ + c0, C_);         // wq[d][c]
        acc = wmma_bf16(A, Bf, acc);
      }
#pragma unroll
      for (int r = 0; r < 8; ++r) {
        const int m = r + hi * 8;
        mt[(size_t)(b * K_ + k0 + m) * C_ + c0 + nn] = (__bf16)acc[r];
      }
    }
  }
}

// ---------------- stage 4: energy -> softmax -> out1/out2 (fused) ----------------
__global__ __launch_bounds__(256) void attn_main(
    const float* __restrict__ x1, const float* __restrict__ x2,
    const __bf16* __restrict__ pv1b, const __bf16* __restrict__ pv2b,
    const __bf16* __restrict__ mt, const float* __restrict__ ebias,
    const float* __restrict__ scale, const float* __restrict__ scale1,
    float* __restrict__ out1, float* __restrict__ out2) {
  __shared__ float xstage[2][32 * 64];  // async-staged x2 chunk [32 c][64 n], dbl-buffered
  __shared__ float eLDS[64][65];        // energy^T tile [k][n], padded
  __shared__ float red[2][4][64];       // partial max / sum
  __shared__ __bf16 attnS[64][72];      // attention [n][k] bf16, padded

  const int b = blockIdx.y;
  const int n0 = blockIdx.x * NT;
  const int t = threadIdx.x;
  const int w = t >> 5, lane = t & 31, hi = lane >> 4, nn = lane & 15;

  // Prefetch residual x1 tile lines into cache while attention is computed.
  for (int c = t; c < C_; c += 256)
    __builtin_prefetch(x1 + (size_t)(b * C_ + c) * N_ + n0, 0, 1);

  // Per-thread async-copy slots: 512 b128 transfers per chunk, 2 per thread.
  const int r0 = t >> 4;               // rows 0..15
  const int r1 = r0 + 16;              // rows 16..31
  const int sg = (t & 15) * 4;         // 4-float (16B) segment within the 64-float row
  const float* gch = x2 + (size_t)b * C_ * N_ + n0 + sg;   // + row*N + chunk*32*N
  const uint32_t l0a = lds_off(&xstage[0][r0 * 64 + sg]);
  const uint32_t l0b = lds_off(&xstage[0][r1 * 64 + sg]);
  const uint32_t l1a = lds_off(&xstage[1][r0 * 64 + sg]);
  const uint32_t l1b = lds_off(&xstage[1][r1 * 64 + sg]);

#define ISSUE_CHUNK(s)                                                          \
  do {                                                                          \
    const float* ga = gch + (size_t)((s) * 32 + r0) * N_;                       \
    const float* gb = gch + (size_t)((s) * 32 + r1) * N_;                       \
    const uint32_t la = ((s) & 1) ? l1a : l0a;                                  \
    const uint32_t lb = ((s) & 1) ? l1b : l0b;                                  \
    asm volatile("global_load_async_to_lds_b128 %0, %1, off"                    \
                 :: "v"(la), "v"(ga) : "memory");                               \
    asm volatile("global_load_async_to_lds_b128 %0, %1, off"                    \
                 :: "v"(lb), "v"(gb) : "memory");                               \
  } while (0)

  // ---- energy^T [64k x 64n] = mt[b] (A, bf16) x x2-tile (B via async LDS stage) ----
  {
    const int k_tile = w & 3;          // waves 0..7: k-strip = w&3, n-halves = w>>2
    const int np = (w >> 2) * 2;
    v8f acc0 = {}, acc1 = {};
    const __bf16* Abase = mt + (size_t)(b * K_ + k_tile * 16) * C_;

    ISSUE_CHUNK(0);
    for (int s = 0; s < C_ / 32; ++s) {
      if (s + 1 < C_ / 32) {
        ISSUE_CHUNK(s + 1);
        asm volatile("s_wait_asynccnt 0x2" ::: "memory");  // chunk s has landed
      } else {
        asm volatile("s_wait_asynccnt 0x0" ::: "memory");
      }
      __syncthreads();                                     // all waves' stages visible
      const float* buf = xstage[s & 1];
      v16bf A  = load_a_bf16(Abase + s * 32, C_);
      v16bf B0 = load_b_stage(buf, np * 16);
      v16bf B1 = load_b_stage(buf, np * 16 + 16);
      acc0 = wmma_bf16(A, B0, acc0);
      acc1 = wmma_bf16(A, B1, acc1);
      __syncthreads();                                     // buffer safe to overwrite
    }
#pragma unroll
    for (int r = 0; r < 8; ++r) {
      const int m = r + hi * 8;
      const float eb = ebias[b * K_ + k_tile * 16 + m];
      eLDS[k_tile * 16 + m][np * 16 + nn]       = acc0[r] + eb;
      eLDS[k_tile * 16 + m][(np + 1) * 16 + nn] = acc1[r] + eb;
    }
  }
  __syncthreads();

  // ---- softmax(|energy|) along k, per column n ----
  {
    const int n = t & 63, seg = t >> 6;
    float v[16];
    float pm = -1e30f;
#pragma unroll
    for (int j = 0; j < 16; ++j) {
      v[j] = fabsf(eLDS[seg * 16 + j][n]);
      pm = fmaxf(pm, v[j]);
    }
    red[0][seg][n] = pm;
    __syncthreads();
    const float gm = fmaxf(fmaxf(red[0][0][n], red[0][1][n]),
                           fmaxf(red[0][2][n], red[0][3][n]));
    float ps = 0.f;
#pragma unroll
    for (int j = 0; j < 16; ++j) { v[j] = __expf(v[j] - gm); ps += v[j]; }
    red[1][seg][n] = ps;
    __syncthreads();
    const float gs = red[1][0][n] + red[1][1][n] + red[1][2][n] + red[1][3][n];
    const float rs = 1.0f / gs;
#pragma unroll
    for (int j = 0; j < 16; ++j) attnS[n][seg * 16 + j] = (__bf16)(v[j] * rs);
  }
  __syncthreads();

  // ---- out = scale * (pv x attn^T) + x_res ; wave w owns c-rows [64w, 64w+64) ----
  const float s1 = scale[0];
  const float s2 = scale1[0];
  for (int sel = 0; sel < 2; ++sel) {
    const __bf16* pv = (sel ? pv2b : pv1b) + (size_t)b * C_ * K_;
    const float* xr = (sel ? x2 : x1) + (size_t)b * C_ * N_;
    float* op = (sel ? out2 : out1) + (size_t)b * C_ * N_;
    const float sc = sel ? s2 : s1;
    for (int ct = 0; ct < 4; ++ct) {
      const int c0 = w * 64 + ct * 16;
      v16bf A0 = load_a_bf16(pv + (size_t)c0 * K_, K_);      // k = 0..31
      v16bf A1 = load_a_bf16(pv + (size_t)c0 * K_ + 32, K_); // k = 32..63
      for (int nt = 0; nt < 4; ++nt) {
        v16bf B0 = load_bt<__bf16>(&attnS[nt * 16][0], 72);  // k = 0..31
        v16bf B1 = load_bt<__bf16>(&attnS[nt * 16][32], 72); // k = 32..63
        v8f acc = {};
        acc = wmma_bf16(A0, B0, acc);
        acc = wmma_bf16(A1, B1, acc);
#pragma unroll
        for (int r = 0; r < 8; ++r) {
          const int m = r + hi * 8;
          const size_t g = (size_t)(c0 + m) * N_ + n0 + nt * 16 + nn;
          op[g] = fmaf(sc, acc[r], xr[g]);
        }
      }
    }
  }
#undef ISSUE_CHUNK
}

// ---------------- host-side launcher ----------------
extern "C" void kernel_launch(void* const* d_in, const int* in_sizes, int n_in,
                              void* d_out, int out_size, void* d_ws, size_t ws_size,
                              hipStream_t stream) {
  (void)in_sizes; (void)n_in; (void)out_size; (void)ws_size;
  const float* x1     = (const float*)d_in[0];
  const float* y1     = (const float*)d_in[1];
  const float* x2     = (const float*)d_in[2];
  const float* y2     = (const float*)d_in[3];
  const float* wq     = (const float*)d_in[4];
  const float* bq     = (const float*)d_in[5];
  const float* wk     = (const float*)d_in[6];
  const float* bk     = (const float*)d_in[7];
  const float* wv     = (const float*)d_in[8];
  const float* bv     = (const float*)d_in[9];
  const float* scale  = (const float*)d_in[10];
  const float* scale1 = (const float*)d_in[11];

  float* out1 = (float*)d_out;
  float* out2 = out1 + (size_t)B_ * C_ * N_;

  // workspace carve-up (~1.8 MB)
  float*  kkf   = (float*)d_ws;                          // [B][K][C4] f32
  float*  ebias = kkf + (size_t)B_ * K_ * C4_;           // [B][K] f32
  __bf16* pv1b  = (__bf16*)(ebias + B_ * K_);            // [B][C][K] bf16
  __bf16* pv2b  = pv1b + (size_t)B_ * C_ * K_;
  __bf16* mt    = pv2b + (size_t)B_ * C_ * K_;           // [B][K][C] bf16

  kk_kernel<<<(B_ * K_ * C4_) / 256, 256, 0, stream>>>(y2, wk, bk, kkf);
  pv_kernel<<<dim3(B_, 2), 256, 0, stream>>>(y1, y2, wv, bv, pv1b, pv2b);
  m_kernel<<<B_, 256, 0, stream>>>(kkf, wq, bq, mt, ebias);
  attn_main<<<dim3(N_ / NT, B_), 256, 0, stream>>>(x1, x2, pv1b, pv2b, mt, ebias,
                                                   scale, scale1, out1, out2);
}
